// VanillaRNN_22720376996312
// MI455X (gfx1250) — matrix operand order repaired and checked by reading, not compiled, not run on hardware
//
#include <hip/hip_runtime.h>
#include <hip/hip_bf16.h>
#include <math.h>

#define VOCAB 32000
#define EMBED 512
#define HID   512
#define BSZ   2
#define TLEN  2048
#define ROWS  (BSZ*TLEN)   // 4096

typedef __attribute__((ext_vector_type(16))) __bf16 v16bf;
typedef __attribute__((ext_vector_type(8)))  float  v8f;
typedef __attribute__((ext_vector_type(4))) unsigned int tdm_u4;
typedef __attribute__((ext_vector_type(8))) int          tdm_i8;
typedef __attribute__((ext_vector_type(4))) int          tdm_i4;

__device__ __forceinline__ unsigned short f2bf(float f) {
  unsigned u = __float_as_uint(f);
  unsigned r = u + 0x7FFFu + ((u >> 16) & 1u);   // round-to-nearest-even
  return (unsigned short)(r >> 16);
}

// ---------------------------------------------------------------------------
// Fragment-order ("swizzled") global layouts, so a flat TDM copy lands
// WMMA-ready data in LDS and each lane reads its fragment as 32 contiguous
// bytes (2x ds_load_b128).
//
// A (MxK bf16), WMMA A 16x32 layout: lane(l<16) row m holds K {0..7,16..23},
// lanes 16..31 hold +8. Chunk unit: [mblock64][kchunk32][mtile4][lane32][elem16]
// B (KxN bf16), WMMA B 32x16 layout: lane = col + 16*((k>>3)&1), elem = (k&7) + 8*(k>>4).
// Chunk unit: [nblock128][kchunk32][ntile8][lane32][elem16]
// ---------------------------------------------------------------------------
__device__ __forceinline__ size_t a_sw_off(int m, int k, int K) {
  const int mb = m >> 6, ml = m & 63, mt = ml >> 4, mr = ml & 15;
  const int kc = k >> 5, kk = k & 31;
  const int sel  = (kk >> 3) & 1;          // 0: K in {0..7,16..23}; 1: +8
  const int kp   = kk - sel * 8;           // {0..7} or {16..23}
  const int elem = (kp < 8) ? kp : kp - 8; // 0..15 within lane
  const int lane = mr + 16 * sel;
  const int nkc = K >> 5;
  return (((size_t)(mb * nkc + kc) * 4 + mt) * 32 + lane) * 16 + elem;
}
__device__ __forceinline__ size_t b_sw_off(int k, int n, int K, int N) {
  const int nb = n >> 7, nl = n & 127, nt = nl >> 4, nc = nl & 15;
  const int kc = k >> 5, kk = k & 31;
  const int lane = nc + 16 * ((kk >> 3) & 1);
  const int elem = (kk & 7) + 8 * (kk >> 4);
  const int nkc = K >> 5;
  return (((size_t)(nb * nkc + kc) * 8 + nt) * 32 + lane) * 16 + elem;
}

// ---------------------------------------------------------------------------
// TDM: async contiguous bf16 block load Global -> LDS (gfx1250 Tensor Data
// Mover). D# packed per CDNA5 ISA. 1D tile,
// data_size=1 (2 bytes), type=2. 6-arg clang-23 builtin.
// ---------------------------------------------------------------------------
__device__ __forceinline__ void tdm_load_block_bf16(
    unsigned lds_off, const void* gptr, unsigned n_elems)
{
  unsigned long long ga = (unsigned long long)gptr;
  tdm_u4 g0;
  g0[0] = 1u;                                   // count=1 (valid)
  g0[1] = lds_off;                              // lds_addr (bytes)
  g0[2] = (unsigned)(ga & 0xFFFFFFFFull);       // global_addr[31:0]
  g0[3] = (unsigned)((ga >> 32) & 0x1FFFFFFull) // global_addr[56:32]
          | (2u << 30);                         // type=2
  const unsigned td0 = n_elems;                 // tensor_dim0
  const unsigned td1 = 1u;                      // tensor_dim1
  tdm_i8 g1;
  g1[0] = (int)(1u << 16);                                  // data_size=1 (2B)
  g1[1] = (int)((td0 & 0xFFFFu) << 16);                     // tensor_dim0[15:0]
  g1[2] = (int)((td0 >> 16) | ((td1 & 0xFFFFu) << 16));     // dim0[31:16]|dim1[15:0]
  g1[3] = (int)((td1 >> 16) | (n_elems << 16));             // dim1[31:16]|tile_dim0
  g1[4] = (int)1;                                           // tile_dim1=1, tile_dim2=0
  g1[5] = (int)n_elems;                                     // tensor_dim0_stride
  g1[6] = 0;
  g1[7] = 0;
  tdm_i4 gz = {0, 0, 0, 0};
  tdm_i8 gz8 = {0, 0, 0, 0, 0, 0, 0, 0};
  __builtin_amdgcn_tensor_load_to_lds(g0, g1, gz, gz, gz8, 0);
}

// ---------------------------------------------------------------------------
// bf16 WMMA GEMM on swizzled operands:
//   C(f32, MxN) = A_sw(bf16, MxK) * B_sw(bf16, KxN) + bias(N)
// Block tile 64x128, K-chunk 32, 256 threads = 8 waves (wave32).
// Wave w: 16-row subtile (w&3), 64-col group (w>>2) -> 4 16x16 accumulators.
// TDM double-buffered LDS staging (wave 0 drives DMA). All fragments are
// fetched into VGPRs first so the 10 ds_load_b128 pipeline ahead of a
// back-to-back WMMA chain.
// ---------------------------------------------------------------------------
#define BM 64
#define BN 128
#define KC 32
#define A_CHUNK (4 * 32 * 16)   // 2048 elems = 4 KB per (mblock,kchunk)
#define B_CHUNK (8 * 32 * 16)   // 4096 elems = 8 KB per (nblock,kchunk)

__global__ __launch_bounds__(256)
void k_gemm_bf16(const unsigned short* __restrict__ A,
                 const unsigned short* __restrict__ Bw,
                 const float* __restrict__ bias,
                 float* __restrict__ C,
                 int N, int K)
{
  __shared__ __attribute__((aligned(32))) unsigned short lA[2][A_CHUNK];
  __shared__ __attribute__((aligned(32))) unsigned short lB[2][B_CHUNK];

  const int tid  = threadIdx.x;
  const int w    = tid >> 5;
  const int lane = tid & 31;
  const int mt   = w & 3;        // 16-row subtile within block
  const int ng   = w >> 2;       // 64-col group within block
  const int mb   = blockIdx.y;
  const int nb   = blockIdx.x;
  const int nkc  = K >> 5;

  const unsigned short* Abase = A + (size_t)mb * nkc * A_CHUNK;
  const unsigned short* Bbase = Bw + (size_t)nb * nkc * B_CHUNK;

  v8f acc[4] = {};

  if (tid < 32) {  // wave 0 issues DMA (scalar-uniform skip for waves 1..7)
    tdm_load_block_bf16((unsigned)(unsigned long long)&lA[0][0], Abase, A_CHUNK);
    tdm_load_block_bf16((unsigned)(unsigned long long)&lB[0][0], Bbase, B_CHUNK);
    __builtin_amdgcn_s_wait_tensorcnt(0);
  }
  __syncthreads();

  for (int kc = 0; kc < nkc; ++kc) {
    const int buf = kc & 1;
    if (tid < 32 && (kc + 1) < nkc) {        // prefetch next chunk, other buffer
      const int nbuf = buf ^ 1;
      tdm_load_block_bf16((unsigned)(unsigned long long)&lA[nbuf][0],
                          Abase + (size_t)(kc + 1) * A_CHUNK, A_CHUNK);
      tdm_load_block_bf16((unsigned)(unsigned long long)&lB[nbuf][0],
                          Bbase + (size_t)(kc + 1) * B_CHUNK, B_CHUNK);
    }

    // Fetch all fragments first (10x ds_load_b128 pipelined), then issue the
    // 4-WMMA chain back-to-back.
    const v16bf af = *(const v16bf*)&lA[buf][(mt * 32 + lane) * 16];
    v16bf bfg[4];
    #pragma unroll
    for (int nt = 0; nt < 4; ++nt)
      bfg[nt] = *(const v16bf*)&lB[buf][((ng * 4 + nt) * 32 + lane) * 16];

    #pragma unroll
    for (int nt = 0; nt < 4; ++nt)
      acc[nt] = __builtin_amdgcn_wmma_f32_16x16x32_bf16(
          false, af, false, bfg[nt], (short)0, acc[nt], false, false);

    if (tid < 32) __builtin_amdgcn_s_wait_tensorcnt(0);
    __syncthreads();
  }

  // C/D layout: lane l, VGPR v -> row v + 8*(l>>4), col l&15.
  const int rsel = (lane >> 4) * 8;
  const int nlo  = lane & 15;
  #pragma unroll
  for (int nt = 0; nt < 4; ++nt) {
    const int gc = nb * BN + ng * 64 + nt * 16 + nlo;
    const float bv = bias[gc];
    #pragma unroll
    for (int v = 0; v < 8; ++v) {
      const int gr = mb * BM + mt * 16 + v + rsel;
      C[(size_t)gr * N + gc] = acc[nt][v] + bv;
    }
  }
}

// ---------------------------------------------------------------------------
// Embedding gather -> swizzled bf16 A rows (row r = b*T + t, flat x layout).
// ---------------------------------------------------------------------------
__global__ __launch_bounds__(256)
void k_embed(const int* __restrict__ x, const float* __restrict__ tab,
             unsigned short* __restrict__ ebf)
{
  const int r = blockIdx.x;
  const int idx = x[r];
  const float* src = tab + (size_t)idx * EMBED;
  for (int e = threadIdx.x; e < EMBED; e += 256)
    ebf[a_sw_off(r, e, EMBED)] = f2bf(src[e]);
}

// f32 (KxN row-major) -> swizzled bf16 B
__global__ void k_cvt_bsw(const float* __restrict__ s, unsigned short* __restrict__ d,
                          int K, int N)
{
  long i = (long)blockIdx.x * blockDim.x + threadIdx.x;
  const long total = (long)K * N;
  const long stride = (long)gridDim.x * blockDim.x;
  for (; i < total; i += stride) {
    const int k = (int)(i / N), n = (int)(i % N);
    d[b_sw_off(k, n, K, N)] = f2bf(s[i]);
  }
}

__global__ void k_init(float* __restrict__ hbufs, unsigned* __restrict__ cnt)
{
  const int i = threadIdx.x;
  if (i == 0) *cnt = 0u;
  for (int k = i; k < 4 * BSZ * HID; k += 256) hbufs[k] = 0.0f;
}

// ---------------------------------------------------------------------------
// Sequential scan. 16 workgroups, each owns 32 hidden columns; weight slices
// (3 x 512x32 f32 = 192 KB) live in LDS (320 KB/WGP). Global arrive/spin
// barrier (monotonic counter) between the two layer phases of each step.
// ---------------------------------------------------------------------------
__device__ __forceinline__ void gsync(unsigned* cnt, unsigned nb)
{
  __syncthreads();
  if (threadIdx.x == 0) {
    __threadfence();
    const unsigned old = atomicAdd(cnt, 1u);
    const unsigned target = (old / nb + 1u) * nb;
    while (atomicAdd(cnt, 0u) < target) { __builtin_amdgcn_s_sleep(1); }
    __threadfence();
  }
  __syncthreads();
}

__global__ __launch_bounds__(256)
void k_scan(const float* __restrict__ X0,      // precomputed x@Wih0+b0, (ROWS,HID)
            const float* __restrict__ Whh0,
            const float* __restrict__ Wih1,
            const float* __restrict__ Whh1,
            const float* __restrict__ b1,
            float* __restrict__ h0a, float* __restrict__ h0b,
            float* __restrict__ h1a, float* __restrict__ h1b,
            unsigned short* __restrict__ outbf, // swizzled A for logits GEMM
            float* __restrict__ dout_h0, float* __restrict__ dout_h1,
            unsigned* __restrict__ bar)
{
  extern __shared__ float smem[];
  float* sWhh0 = smem;                       // 512*32
  float* sWih1 = sWhh0 + HID * 32;
  float* sWhh1 = sWih1 + HID * 32;
  float* sH    = sWhh1 + HID * 32;           // 2*512 staging
  float* sH2   = sH + BSZ * HID;             // 2*512 staging

  const int tid = threadIdx.x;
  const int jg  = blockIdx.x * 32;           // owned column group

  for (int i = tid; i < HID * 32; i += 256) {
    const int k = i >> 5, j = i & 31;
    sWhh0[i] = Whh0[(size_t)k * HID + jg + j];
    sWih1[i] = Wih1[(size_t)k * HID + jg + j];
    sWhh1[i] = Whh1[(size_t)k * HID + jg + j];
  }
  __syncthreads();

  const int out  = tid >> 2, part = tid & 3; // 4 lanes per output, 64 outputs
  const int b    = out >> 5, j = out & 31;
  const int k0   = part * 128;

  float* h0c = h0a; float* h0n = h0b;
  float* h1c = h1a; float* h1n = h1b;

  for (int t = 0; t < TLEN; ++t) {
    // ---- layer 0: h0n = tanh(X0[t] + h0 @ Whh0) ----
    for (int i = tid; i < BSZ * HID; i += 256) sH[i] = h0c[i];
    __syncthreads();
    {
      float acc = 0.0f;
      const float* hv = &sH[b * HID + k0];
      #pragma unroll 4
      for (int k = 0; k < 128; ++k)
        acc += hv[k] * sWhh0[(k0 + k) * 32 + j];
      acc += __shfl_xor(acc, 1, 32);
      acc += __shfl_xor(acc, 2, 32);
      if (part == 0) {
        const size_t r = (size_t)b * TLEN + t;
        h0n[b * HID + jg + j] = tanhf(X0[r * HID + jg + j] + acc);
      }
    }
    __threadfence();
    gsync(bar, 16);

    // ---- layer 1: h1n = tanh(h0n @ Wih1 + h1 @ Whh1 + b1) ----
    for (int i = tid; i < BSZ * HID; i += 256) { sH[i] = h0n[i]; sH2[i] = h1c[i]; }
    __syncthreads();
    {
      float a1 = 0.0f, a2 = 0.0f;
      const float* hv1 = &sH[b * HID + k0];
      const float* hv2 = &sH2[b * HID + k0];
      #pragma unroll 4
      for (int k = 0; k < 128; ++k) {
        a1 += hv1[k] * sWih1[(k0 + k) * 32 + j];
        a2 += hv2[k] * sWhh1[(k0 + k) * 32 + j];
      }
      float acc = a1 + a2;
      acc += __shfl_xor(acc, 1, 32);
      acc += __shfl_xor(acc, 2, 32);
      if (part == 0) {
        const float v = tanhf(acc + b1[jg + j]);
        h1n[b * HID + jg + j] = v;
        const int r = b * TLEN + t;
        outbf[a_sw_off(r, jg + j, HID)] = f2bf(v);
      }
    }
    __threadfence();
    gsync(bar, 16);

    float* tp;
    tp = h0c; h0c = h0n; h0n = tp;
    tp = h1c; h1c = h1n; h1n = tp;
  }

  // final states (each WG wrote its own columns)
  if (part == 0) {
    dout_h0[b * HID + jg + j] = h0c[b * HID + jg + j];
    dout_h1[b * HID + jg + j] = h1c[b * HID + jg + j];
  }
}

// ---------------------------------------------------------------------------
extern "C" void kernel_launch(void* const* d_in, const int* in_sizes, int n_in,
                              void* d_out, int out_size, void* d_ws, size_t ws_size,
                              hipStream_t stream)
{
  const int*   x     = (const int*)  d_in[0];
  const float* emb_t = (const float*)d_in[1];
  const float* Wih0  = (const float*)d_in[2];
  const float* Whh0  = (const float*)d_in[3];
  const float* b0    = (const float*)d_in[4];
  const float* Wih1  = (const float*)d_in[5];
  const float* Whh1  = (const float*)d_in[6];
  const float* b1    = (const float*)d_in[7];
  const float* Wout  = (const float*)d_in[8];
  const float* bout  = (const float*)d_in[9];
  float* dout = (float*)d_out;

  char* ws = (char*)d_ws;
  size_t off = 0;
  auto take = [&](size_t n) { size_t a = off; off += (n + 255) & ~(size_t)255; return a; };
  unsigned short* Wout_bf = (unsigned short*)(ws + take((size_t)HID * VOCAB * 2));
  unsigned short* Wih0_bf = (unsigned short*)(ws + take((size_t)EMBED * HID * 2));
  unsigned short* emb_bf  = (unsigned short*)(ws + take((size_t)ROWS * EMBED * 2));
  float*          X0      = (float*)         (ws + take((size_t)ROWS * HID * 4));
  unsigned short* out_bf  = (unsigned short*)(ws + take((size_t)ROWS * HID * 2));
  float*          hb      = (float*)         (ws + take((size_t)4 * BSZ * HID * 4));
  unsigned*       bar     = (unsigned*)      (ws + take(256));

  k_init<<<1, 256, 0, stream>>>(hb, bar);
  k_cvt_bsw<<<512, 256, 0, stream>>>(Wout, Wout_bf, HID, VOCAB);
  k_cvt_bsw<<<64, 256, 0, stream>>>(Wih0, Wih0_bf, EMBED, HID);
  k_embed<<<ROWS, 256, 0, stream>>>(x, emb_t, emb_bf);

  // X0 = emb @ Wih0 + b0  (4096 x 512 x 512, WMMA bf16)
  dim3 g1(HID / BN, ROWS / BM);
  k_gemm_bf16<<<g1, 256, 0, stream>>>(emb_bf, Wih0_bf, b0, X0, HID, EMBED);

  // sequential 2-layer scan (16 WGs, 192 KB weight slices + staging in LDS)
  const size_t shmem = (size_t)(3 * HID * 32 + 2 * BSZ * HID) * sizeof(float);
  const size_t OFFH = (size_t)ROWS * VOCAB;
  k_scan<<<16, 256, shmem, stream>>>(X0, Whh0, Wih1, Whh1, b1,
                                     hb, hb + BSZ * HID,
                                     hb + 2 * BSZ * HID, hb + 3 * BSZ * HID,
                                     out_bf, dout + OFFH, dout + OFFH + BSZ * HID, bar);

  // logits = out @ Wout + bout  (4096 x 32000 x 512, WMMA bf16) -> dominant GEMM
  dim3 g2(VOCAB / BN, ROWS / BM);
  k_gemm_bf16<<<g2, 256, 0, stream>>>(out_bf, Wout_bf, bout, dout, ROWS, VOCAB);
}
